// anchordetection3dhead_24369644438187
// MI455X (gfx1250) — matrix operand order repaired and checked by reading, not compiled, not run on hardware
//
#include <hip/hip_runtime.h>

// ---------------------------------------------------------------------------
// AnchorDetection3D head on MI455X (gfx1250, wave32).
// Compute-bound (~1 TFLOP vs ~10us of HBM traffic) -> all convs lowered to
// implicit GEMM on v_wmma_f32_16x16x32_bf16 with f32 accumulation.
// Activations kept in bf16 [C][B][H+2][W+2] zero-padded layout so 3x3 im2col
// is a pure pointer shift; epilogues fused (bias/ReLU/BN/NHWC-flatten).
// LDS tiles are stored fragment-ready: A as [m][k], B as [n][k], so both
// fragment loads are plain ds_load_b128 pairs with no VALU repacking.
// A-tile global->LDS uses GLOBAL_LOAD_ASYNC_TO_LDS_B128 (ASYNCcnt) -- builtin
// confirmed present; falls back to VGPR-staged copies if absent.
// ---------------------------------------------------------------------------

typedef __attribute__((ext_vector_type(16))) __bf16 v16bf;
typedef __attribute__((ext_vector_type(8)))  float  v8f;
typedef int v4i_ __attribute__((vector_size(16)));   // builtin's int4 pointee type

#if defined(__has_builtin)
#  if __has_builtin(__builtin_amdgcn_global_load_async_to_lds_b128) && \
      __has_builtin(__builtin_amdgcn_s_wait_asynccnt)
#    define USE_ASYNC_LDS 1
#  endif
#endif
#ifndef USE_ASYNC_LDS
#  define USE_ASYNC_LDS 0
#endif

#define CIN    1024
#define FEAT   1024
#define KTOT   9216          // 9 * 1024
#define NKS    288           // KTOT / 32
#define HH     18
#define WW     80
#define BB_    8
#define HP     20            // H + 2
#define WP     82            // W + 2
#define PB     1640          // HP*WP, padded cols per batch
#define NP     13120         // B*PB, total padded GEMM columns
#define NT     103           // ceil(NP/128)
#define BM     128
#define BN     128
#define BK     32
#define KP     40            // LDS k-row pitch (u16): 80B = 5*16B, aligned
#define SAMPST 1664          // row stride of sample chunk (13*128)
#define OC_CLS 96
#define OC_REG 416

__device__ __forceinline__ unsigned short f2bf(float f) {
    unsigned u = __float_as_uint(f);
    unsigned r = u + 0x7FFFu + ((u >> 16) & 1u);
    return (unsigned short)(r >> 16);
}
__device__ __forceinline__ float bf2f(unsigned short h) {
    return __uint_as_float(((unsigned)h) << 16);
}

union StU {                       // 32-byte global<->LDS staging
    uint4 q[2];
    unsigned u[8];
    unsigned short s[16];
};
union Frag {                      // one WMMA A/B fragment (16 bf16 per lane)
    uint4 q[2];
    unsigned u[8];
    v16bf v;
};

// ---------------------------------------------------------------------------
// fp32 -> bf16 conversions
// ---------------------------------------------------------------------------
__global__ void __launch_bounds__(256) feat_to_act(const float* __restrict__ in,
                                                   unsigned short* __restrict__ act) {
    size_t i = (size_t)blockIdx.x * 256 + threadIdx.x;
    if (i >= (size_t)CIN * NP) return;
    int c = (int)(i / NP);
    int p = (int)(i - (size_t)c * NP);
    int b = p / PB;  int rem = p - b * PB;
    int yp = rem / WP; int xp = rem - yp * WP;
    float v = 0.f;
    if (yp > 0 && yp < HP - 1 && xp > 0 && xp < WP - 1)
        v = in[(((size_t)b * CIN + c) * HH + (yp - 1)) * WW + (xp - 1)];
    act[i] = f2bf(v);
}

// w[oc][ic][ky][kx] fp32  ->  bf16 [Mpad][kk*CIN + ic], zero-filled past OC
__global__ void __launch_bounds__(256) weight_to_bf16(const float* __restrict__ w,
                                                      unsigned short* __restrict__ out,
                                                      int OC, int Mpad) {
    size_t i = (size_t)blockIdx.x * 256 + threadIdx.x;
    if (i >= (size_t)Mpad * KTOT) return;
    int oc = (int)(i / KTOT);
    int k  = (int)(i - (size_t)oc * KTOT);
    int kk = k >> 10;
    int ic = k & 1023;
    float v = (oc < OC) ? w[((size_t)oc * CIN + ic) * 9 + kk] : 0.f;
    out[i] = f2bf(v);
}

// ---------------------------------------------------------------------------
// Implicit-GEMM conv kernel. 256 threads = 8 waves, tile 128x128x32,
// double-buffered LDS, bf16 WMMA 16x16x32, f32 accumulation.
//   EPI 0: bias + ReLU            -> bf16 act (border columns forced to 0)
//   EPI 1: bias + BN + ReLU       -> bf16 act (border columns forced to 0)
//   EPI 2: bias                   -> fp32 NHWC out (== anchor_flatten layout)
//   EPI 3: bias                   -> fp32 [oc][NP]  (offset conv)
//   SHIFT: 3x3 im2col via padded-plane pointer shift; else direct rows.
// ---------------------------------------------------------------------------
template<int EPI, bool SHIFT>
__global__ void __launch_bounds__(256)
gemm_conv(const unsigned short* __restrict__ A,     // [Mpad][KTOT] bf16 weights
          const unsigned short* __restrict__ Bact,  // act base (ch stride NP) or samp (stride bstride)
          int M, int ncols, int bstride, int outColOff,
          const float* __restrict__ bias,
          const float* __restrict__ bng, const float* __restrict__ bnb,
          unsigned short* __restrict__ outAct, float* __restrict__ outF) {
    __shared__ unsigned short As[2][BM][KP];       // [m][k], 80B rows
    __shared__ unsigned short Bs[2][BN][KP];       // [n][k], fragment-ready

    const int tid  = threadIdx.x;
    const int wave = tid >> 5, lane = tid & 31;
    const int wm = wave & 1, wn = wave >> 1;       // 2x4 wave grid: 64x32 per wave
    const int mtile = blockIdx.y * BM;
    const int ntile = blockIdx.x * BN;

    const int amr = tid >> 1, aks = (tid & 1) * 16;   // A loader: row, k-offset
    const int bnc = tid & 127;                        // B loader: n column
    const int bkh = (tid >> 7) * 16;                  // B loader: k-half

    v8f acc[4][2];
#pragma unroll
    for (int sm = 0; sm < 4; ++sm)
#pragma unroll
        for (int sn = 0; sn < 2; ++sn)
#pragma unroll
            for (int v = 0; v < 8; ++v) acc[sm][sn][v] = 0.f;

    const unsigned short* Arow = A + (size_t)(mtile + amr) * KTOT + aks;

    // ---- A tile: async global->LDS when available, else staged ----
    auto aload = [&](int ks, int buf, StU& sa) {
        const unsigned short* pa = Arow + ks * BK;
#if USE_ASYNC_LDS
        unsigned short* lp = &As[buf][amr][aks];
        __builtin_amdgcn_global_load_async_to_lds_b128(
            (__attribute__((address_space(1))) v4i_*)pa,
            (__attribute__((address_space(3))) v4i_*)lp, 0, 0);
        __builtin_amdgcn_global_load_async_to_lds_b128(
            (__attribute__((address_space(1))) v4i_*)(pa + 8),
            (__attribute__((address_space(3))) v4i_*)(lp + 8), 0, 0);
        (void)sa;
#else
        (void)buf;
        sa.q[0] = *(const uint4*)pa;
        sa.q[1] = *(const uint4*)(pa + 8);
#endif
    };
    auto astore = [&](int buf, const StU& sa) {
#if !USE_ASYNC_LDS
        *(uint4*)&As[buf][amr][aks]     = sa.q[0];
        *(uint4*)&As[buf][amr][aks + 8] = sa.q[1];
#else
        (void)buf; (void)sa;
#endif
    };
    // ---- B tile: k-strided coalesced u16 gathers, staged in VGPRs ----
    auto bload = [&](int ks, StU& sb) {
        if (SHIFT) {
            int kk = ks >> 5;                       // 32 k-steps per (ky,kx)
            int c0 = (ks & 31) * BK + bkh;
            int sh = (kk / 3 - 1) * WP + (kk % 3 - 1);
            const unsigned short* pb = Bact + (size_t)c0 * NP + (ntile + bnc + sh);
#pragma unroll
            for (int i = 0; i < 16; ++i) sb.s[i] = pb[(size_t)i * NP];
        } else {
            int k0 = ks * BK + bkh;
            const unsigned short* pb = Bact + (size_t)k0 * bstride + ntile + bnc;
#pragma unroll
            for (int i = 0; i < 16; ++i) sb.s[i] = pb[(size_t)i * bstride];
        }
    };
    auto bstore = [&](int buf, const StU& sb) {
        *(uint4*)&Bs[buf][bnc][bkh]     = sb.q[0];
        *(uint4*)&Bs[buf][bnc][bkh + 8] = sb.q[1];
    };

    {
        StU sa, sb;
        aload(0, 0, sa);
        bload(0, sb);
        astore(0, sa);
        bstore(0, sb);
    }
#if USE_ASYNC_LDS
    __builtin_amdgcn_s_wait_asynccnt(0);
#endif
    __syncthreads();

    const int mrow  = lane & 15;
    const int khalf = lane >> 4;

    for (int ks = 0; ks < NKS; ++ks) {
        const int cur = ks & 1;
        const bool more = (ks + 1) < NKS;
        StU na, nb;
        if (more) {
            aload(ks + 1, (ks + 1) & 1, na);   // async: lands in LDS directly
            bload(ks + 1, nb);
        }

        Frag af[4], bf[2];
#pragma unroll
        for (int sm = 0; sm < 4; ++sm) {
            const unsigned short* ap = &As[cur][wm * 64 + sm * 16 + mrow][khalf * 8];
            af[sm].q[0] = *(const uint4*)ap;
            af[sm].q[1] = *(const uint4*)(ap + 16);
        }
#pragma unroll
        for (int sn = 0; sn < 2; ++sn) {
            const unsigned short* bp = &Bs[cur][wn * 32 + sn * 16 + mrow][khalf * 8];
            bf[sn].q[0] = *(const uint4*)bp;
            bf[sn].q[1] = *(const uint4*)(bp + 16);
        }
#pragma unroll
        for (int sm = 0; sm < 4; ++sm)
#pragma unroll
            for (int sn = 0; sn < 2; ++sn)
                acc[sm][sn] = __builtin_amdgcn_wmma_f32_16x16x32_bf16(
                    false, af[sm].v, false, bf[sn].v, (short)0, acc[sm][sn],
                    false, false);

        if (more) {
            astore((ks + 1) & 1, na);
            bstore((ks + 1) & 1, nb);
        }
#if USE_ASYNC_LDS
        if (more) __builtin_amdgcn_s_wait_asynccnt(0);
#endif
        __syncthreads();
    }

    // ---------------- epilogue ----------------
    const int vrow  = lane >> 4;
    const int ncol0 = lane & 15;
#pragma unroll
    for (int sm = 0; sm < 4; ++sm) {
#pragma unroll
        for (int sn = 0; sn < 2; ++sn) {
            int j = ntile + wn * 32 + sn * 16 + ncol0;
            if (j >= ncols) continue;
            int p  = outColOff + j;
            int b  = p / PB;  int rem = p - b * PB;
            int yp = rem / WP; int xp = rem - yp * WP;
            bool border = (yp == 0) | (yp == HP - 1) | (xp == 0) | (xp == WP - 1);
#pragma unroll
            for (int v = 0; v < 8; ++v) {
                int oc = mtile + wm * 64 + sm * 16 + vrow * 8 + v;
                if (oc >= M) continue;
                float val = acc[sm][sn][v] + bias[oc];
                if constexpr (EPI == 0) {
                    val = border ? 0.f : fmaxf(val, 0.f);
                    outAct[(size_t)oc * NP + p] = f2bf(val);
                } else if constexpr (EPI == 1) {
                    float sc = bng[oc] * rsqrtf(1.f + 1e-5f);
                    val = val * sc + bnb[oc];
                    val = border ? 0.f : fmaxf(val, 0.f);
                    outAct[(size_t)oc * NP + p] = f2bf(val);
                } else if constexpr (EPI == 2) {
                    if (!border) {
                        int nr = (b * HH + (yp - 1)) * WW + (xp - 1);
                        outF[(size_t)nr * M + oc] = val;
                    }
                } else {
                    outF[(size_t)oc * NP + p] = val;
                }
            }
        }
    }
}

// ---------------------------------------------------------------------------
// Modulated deformable sampler: bilinear gather * sigmoid(mask) into
// samp[(kk*1024+c)][local padded pixel], one batch per set of launches.
// grid (12, 72): x = 128-pixel block, y = kk(9) * channel-group(8), block 128.
// ---------------------------------------------------------------------------
__global__ void __launch_bounds__(128)
mdcn_sampler(const unsigned short* __restrict__ act, const float* __restrict__ off,
             unsigned short* __restrict__ samp, int b) {
    int tid = threadIdx.x;
    int pix = blockIdx.x * 128 + tid;
    int kk  = blockIdx.y % 9;
    int cg  = blockIdx.y / 9;
    if (pix >= HH * WW) return;
    int y = pix / WW, x = pix - (pix / WW) * WW;
    int pg = b * PB + (y + 1) * WP + (x + 1);

    float dy = off[(size_t)kk * NP + pg];
    float dx = off[(size_t)(9 + kk) * NP + pg];
    float ml = off[(size_t)(18 + kk) * NP + pg];
    float m  = 1.f / (1.f + __expf(-ml));

    float py = (float)y - 1.f + (float)(kk / 3) + dy;
    float px = (float)x - 1.f + (float)(kk % 3) + dx;
    float y0f = floorf(py), x0f = floorf(px);
    int y0 = (int)y0f, x0 = (int)x0f;
    float wy1 = py - y0f, wx1 = px - x0f;
    float wys[2] = {1.f - wy1, wy1};
    float wxs[2] = {1.f - wx1, wx1};
    int ys[2] = {y0, y0 + 1}, xs[2] = {x0, x0 + 1};

    int idx[4]; float wgt[4];
#pragma unroll
    for (int a2 = 0; a2 < 2; ++a2)
#pragma unroll
        for (int b2 = 0; b2 < 2; ++b2) {
            int yy = ys[a2], xx = xs[b2];
            bool vld = (yy >= 0) && (yy < HH) && (xx >= 0) && (xx < WW);
            int yc = yy < 0 ? 0 : (yy > HH - 1 ? HH - 1 : yy);
            int xc = xx < 0 ? 0 : (xx > WW - 1 ? WW - 1 : xx);
            idx[a2 * 2 + b2] = b * PB + (yc + 1) * WP + (xc + 1);
            wgt[a2 * 2 + b2] = vld ? wys[a2] * wxs[b2] * m : 0.f;  // mask folded in
        }
    int lp = (y + 1) * WP + (x + 1);
    for (int ci = 0; ci < 128; ++ci) {
        int c = cg * 128 + ci;
        const unsigned short* ap = act + (size_t)c * NP;
        float v = wgt[0] * bf2f(ap[idx[0]]) + wgt[1] * bf2f(ap[idx[1]])
                + wgt[2] * bf2f(ap[idx[2]]) + wgt[3] * bf2f(ap[idx[3]]);
        samp[(size_t)(kk * CIN + c) * SAMPST + lp] = f2bf(v);
    }
}

// ---------------------------------------------------------------------------
extern "C" void kernel_launch(void* const* d_in, const int* in_sizes, int n_in,
                              void* d_out, int out_size, void* d_ws, size_t ws_size,
                              hipStream_t stream) {
    (void)in_sizes; (void)n_in; (void)out_size; (void)ws_size;
    const float* feat   = (const float*)d_in[0];
    const float* cls_w1 = (const float*)d_in[1];  const float* cls_b1 = (const float*)d_in[2];
    const float* cls_w2 = (const float*)d_in[3];  const float* cls_b2 = (const float*)d_in[4];
    const float* cls_w3 = (const float*)d_in[5];  const float* cls_b3 = (const float*)d_in[6];
    const float* off_w  = (const float*)d_in[7];  const float* off_b  = (const float*)d_in[8];
    const float* dcn_w  = (const float*)d_in[9];  const float* dcn_b  = (const float*)d_in[10];
    const float* bn1_g  = (const float*)d_in[11]; const float* bn1_b  = (const float*)d_in[12];
    const float* reg_w2 = (const float*)d_in[13]; const float* reg_b2 = (const float*)d_in[14];
    const float* bn2_g  = (const float*)d_in[15]; const float* bn2_b  = (const float*)d_in[16];
    const float* reg_w3 = (const float*)d_in[17]; const float* reg_b3 = (const float*)d_in[18];

    char* ws = (char*)d_ws;
    size_t cur = 0;
    auto alloc = [&](size_t bytes) { size_t o = cur; cur = (cur + bytes + 255) & ~(size_t)255; return o; };

    const size_t ACT_DATA  = (size_t)CIN * NP;                 // bf16 elems
    const size_t ACT_ALLOC = (128 + ACT_DATA + 512) * 2;       // + front/back guards (im2col shifts)
    size_t oActF = alloc(ACT_ALLOC);
    size_t oH1   = alloc(ACT_ALLOC);                           // h1 / r1 (reused)
    size_t oH2   = alloc(ACT_ALLOC);                           // h2 / r2 (reused)
    size_t oOff  = alloc((size_t)27 * NP * 4);                 // offset-conv fp32
    size_t oSmp  = alloc((size_t)KTOT * SAMPST * 2);           // per-batch sample chunk
    size_t oWc1  = alloc((size_t)1024 * KTOT * 2);
    size_t oWc2  = alloc((size_t)1024 * KTOT * 2);
    size_t oWc3  = alloc((size_t)128  * KTOT * 2);
    size_t oWof  = alloc((size_t)128  * KTOT * 2);
    size_t oWdc  = alloc((size_t)1024 * KTOT * 2);
    size_t oWr2  = alloc((size_t)1024 * KTOT * 2);
    size_t oWr3  = alloc((size_t)512  * KTOT * 2);             // total ~205 MB

    unsigned short* actF = (unsigned short*)(ws + oActF) + 128;
    unsigned short* h1   = (unsigned short*)(ws + oH1)   + 128;
    unsigned short* h2   = (unsigned short*)(ws + oH2)   + 128;
    float*          offF = (float*)(ws + oOff);
    unsigned short* smp  = (unsigned short*)(ws + oSmp);
    unsigned short* wC1  = (unsigned short*)(ws + oWc1);
    unsigned short* wC2  = (unsigned short*)(ws + oWc2);
    unsigned short* wC3  = (unsigned short*)(ws + oWc3);
    unsigned short* wOf  = (unsigned short*)(ws + oWof);
    unsigned short* wDc  = (unsigned short*)(ws + oWdc);
    unsigned short* wR2  = (unsigned short*)(ws + oWr2);
    unsigned short* wR3  = (unsigned short*)(ws + oWr3);

    float* outCls = (float*)d_out;
    float* outReg = outCls + (size_t)BB_ * HH * WW * OC_CLS;

    // zero act buffers (guards + borders)
    (void)hipMemsetAsync(ws + oActF, 0, ACT_ALLOC, stream);
    (void)hipMemsetAsync(ws + oH1,   0, ACT_ALLOC, stream);
    (void)hipMemsetAsync(ws + oH2,   0, ACT_ALLOC, stream);

    // conversions
    {
        int g = (int)(((size_t)CIN * NP + 255) / 256);
        feat_to_act<<<g, 256, 0, stream>>>(feat, actF);
    }
    auto cvtW = [&](const float* src, unsigned short* dst, int OC, int Mpad) {
        int g = (int)(((size_t)Mpad * KTOT + 255) / 256);
        weight_to_bf16<<<g, 256, 0, stream>>>(src, dst, OC, Mpad);
    };
    cvtW(cls_w1, wC1, 1024, 1024);
    cvtW(cls_w2, wC2, 1024, 1024);
    cvtW(cls_w3, wC3, OC_CLS, 128);
    cvtW(off_w,  wOf, 27, 128);
    cvtW(dcn_w,  wDc, 1024, 1024);
    cvtW(reg_w2, wR2, 1024, 1024);
    cvtW(reg_w3, wR3, OC_REG, 512);

    // offset conv: features -> 27 offset/mask channels (fp32)
    gemm_conv<3, true><<<dim3(NT, 1), 256, 0, stream>>>(
        wOf, actF, 27, NP, 0, 0, off_b, nullptr, nullptr, nullptr, offF);

    // classification branch
    gemm_conv<0, true><<<dim3(NT, 8), 256, 0, stream>>>(
        wC1, actF, 1024, NP, 0, 0, cls_b1, nullptr, nullptr, h1, nullptr);
    gemm_conv<0, true><<<dim3(NT, 8), 256, 0, stream>>>(
        wC2, h1, 1024, NP, 0, 0, cls_b2, nullptr, nullptr, h2, nullptr);
    gemm_conv<2, true><<<dim3(NT, 1), 256, 0, stream>>>(
        wC3, h2, OC_CLS, NP, 0, 0, cls_b3, nullptr, nullptr, nullptr, outCls);

    // regression branch: MDCN (sample per batch, then dense GEMM) + BN1 + ReLU
    for (int b = 0; b < BB_; ++b) {
        mdcn_sampler<<<dim3(12, 72), 128, 0, stream>>>(actF, offF, smp, b);
        gemm_conv<1, false><<<dim3(13, 8), 256, 0, stream>>>(
            wDc, smp, 1024, PB, SAMPST, b * PB, dcn_b, bn1_g, bn1_b, h1, nullptr);
    }
    gemm_conv<1, true><<<dim3(NT, 8), 256, 0, stream>>>(
        wR2, h1, 1024, NP, 0, 0, reg_b2, bn2_g, bn2_b, h2, nullptr);
    gemm_conv<2, true><<<dim3(NT, 4), 256, 0, stream>>>(
        wR3, h2, OC_REG, NP, 0, 0, reg_b3, nullptr, nullptr, nullptr, outReg);
}
